// ResnetGru_65008624992699
// MI455X (gfx1250) — matrix operand order repaired and checked by reading, not compiled, not run on hardware
//
#include <hip/hip_runtime.h>
#include <math.h>

// ---------------------------------------------------------------------------
// Types for CDNA5 WMMA (wave32): D(16x16 f32) = A(16x32 bf16) x B(32x16 bf16) + C
// ---------------------------------------------------------------------------
typedef __bf16 bf16_t;
typedef __attribute__((ext_vector_type(16))) __bf16 v16bf;
typedef __attribute__((ext_vector_type(8)))  __bf16 v8bf;
typedef __attribute__((ext_vector_type(8)))  float  v8f;

__device__ __forceinline__ float bf2f(bf16_t x) { return (float)x; }
__device__ __forceinline__ bf16_t f2bf(float f) { return (bf16_t)f; }

__device__ __forceinline__ v8f wmma_bf16(v16bf a, v16bf b, v8f c) {
  // 8 args: (neg_a, A, neg_b, B, c_mod, C, reuse_a, reuse_b)
  return __builtin_amdgcn_wmma_f32_16x16x32_bf16(false, a, false, b, (short)0, c,
                                                 false, false);
}

// A-fragment (16x32 bf16) loaded from a row-major [.., K] matrix.
// lane l: row = l&15, khalf = l>>4; elements = src[k0..k0+8) ++ src[k0+16..k0+24)
// with k0 = kb*32 + khalf*8  (matches ISA 16-bit A layout).
__device__ __forceinline__ v16bf load_afrag(const bf16_t* __restrict__ p) {
  v8bf lo = *(const v8bf*)(p);
  v8bf hi = *(const v8bf*)(p + 16);
  return __builtin_shufflevector(lo, hi, 0,1,2,3,4,5,6,7,8,9,10,11,12,13,14,15);
}

// ---------------------------------------------------------------------------
// Weight packing: B-fragments stored contiguously per (ntile, kblock, lane).
// B 32x16 bf16 layout: lane l -> column n = l&15; element e -> K = (l>>4)*8 + (e&7) + (e>>3)*16
// within the 32-wide K-block.  Linear: ((nt*Kb + kb)*32 + lane)*16 + e
// ---------------------------------------------------------------------------
__global__ void pack_dense(const float* __restrict__ W, bf16_t* __restrict__ out,
                           int K, int Npad, int Nsrc) { // W: [Nsrc][K] row-major
  long idx = (long)blockIdx.x * blockDim.x + threadIdx.x;
  long total = (long)Npad * K;
  if (idx >= total) return;
  int e    = (int)(idx & 15);
  int lane = (int)((idx >> 4) & 31);
  long q   = idx >> 9;
  int Kb   = K >> 5;
  int kb   = (int)(q % Kb);
  int nt   = (int)(q / Kb);
  int n    = nt * 16 + (lane & 15);
  int kk   = kb * 32 + (lane >> 4) * 8 + (e & 7) + (e >> 3) * 16;
  float v  = (n < Nsrc) ? W[(long)n * K + kk] : 0.f;
  out[idx] = f2bf(v);
}

__global__ void pack_conv(const float* __restrict__ W, bf16_t* __restrict__ out,
                          int Cout, int Cin, int KH, int KW) { // W: OIHW
  long idx  = (long)blockIdx.x * blockDim.x + threadIdx.x;
  long Ktot = (long)KH * KW * Cin;
  long total = Cout * Ktot;
  if (idx >= total) return;
  int e    = (int)(idx & 15);
  int lane = (int)((idx >> 4) & 31);
  long q   = idx >> 9;
  long Kb  = Ktot >> 5;
  int kb   = (int)(q % Kb);
  int nt   = (int)(q / Kb);
  int n    = nt * 16 + (lane & 15);
  int kk   = kb * 32 + (lane >> 4) * 8 + (e & 7) + (e >> 3) * 16; // Cin % 32 == 0 -> block within one tap
  int tap  = kk / Cin, c = kk % Cin;
  int ky   = tap / KW, kx = tap % KW;
  out[idx] = f2bf(W[(((long)n * Cin + c) * KH + ky) * KW + kx]);
}

// ---------------------------------------------------------------------------
// Stem: 7x7/2 conv (Cin=3, not WMMA-friendly) + BN + ReLU.  NCHW f32 -> NHWC bf16
// ---------------------------------------------------------------------------
__global__ void stem_conv(const float* __restrict__ img, const float* __restrict__ W,
                          const float* __restrict__ s, const float* __restrict__ b,
                          bf16_t* __restrict__ out) { // out [16,112,112,64]
  long idx = (long)blockIdx.x * blockDim.x + threadIdx.x;
  const long total = 16L * 112 * 112 * 64;
  if (idx >= total) return;
  int c = (int)(idx & 63);
  long q = idx >> 6;
  int ow = (int)(q % 112); q /= 112;
  int oh = (int)(q % 112); q /= 112;
  int n = (int)q;
  int ih0 = oh * 2 - 3, iw0 = ow * 2 - 3;
  float acc = 0.f;
  for (int ci = 0; ci < 3; ++ci)
    for (int ky = 0; ky < 7; ++ky) {
      int ih = ih0 + ky;
      if ((unsigned)ih >= 224u) continue;
      const float* ip = img + (((long)n * 3 + ci) * 224 + ih) * 224;
      const float* wp = W + (((long)c * 3 + ci) * 7 + ky) * 7;
      for (int kx = 0; kx < 7; ++kx) {
        int iw = iw0 + kx;
        if ((unsigned)iw >= 224u) continue;
        acc += ip[iw] * wp[kx];
      }
    }
  out[idx] = f2bf(fmaxf(acc * s[c] + b[c], 0.f));
}

__global__ void maxpool3x3s2(const bf16_t* __restrict__ in, bf16_t* __restrict__ out) {
  long idx = (long)blockIdx.x * blockDim.x + threadIdx.x;
  const long total = 16L * 56 * 56 * 64;
  if (idx >= total) return;
  int c = (int)(idx & 63);
  long q = idx >> 6;
  int ow = (int)(q % 56); q /= 56;
  int oh = (int)(q % 56);
  int n  = (int)(q / 56);
  int ih0 = oh * 2 - 1, iw0 = ow * 2 - 1;
  float m = -INFINITY;
  for (int ky = 0; ky < 3; ++ky) {
    int ih = ih0 + ky;
    if ((unsigned)ih >= 112u) continue;
    for (int kx = 0; kx < 3; ++kx) {
      int iw = iw0 + kx;
      if ((unsigned)iw >= 112u) continue;
      m = fmaxf(m, bf2f(in[(((long)n * 112 + ih) * 112 + iw) * 64 + c]));
    }
  }
  out[idx] = f2bf(m);
}

// ---------------------------------------------------------------------------
// Implicit-GEMM conv + folded BN (+residual, +ReLU).  One wave per 16x16 tile.
// B-fragments are streamed through LDS with the CDNA5 async-to-LDS pipeline:
// chunk = 2 K-blocks (2 KB), double-buffered; ASYNCcnt gates the consumer,
// DScnt closes the WAR hazard before overwriting a buffer.
// ---------------------------------------------------------------------------
__global__ __launch_bounds__(32) void conv_bn_wmma(
    const bf16_t* __restrict__ in, const bf16_t* __restrict__ wpack,
    const float* __restrict__ scale, const float* __restrict__ bias,
    const bf16_t* __restrict__ resid, bf16_t* __restrict__ out,
    int NB, int IH, int IW, int Cin, int OH, int OW, int Cout,
    int KH, int KW, int stride, int pad, int relu) {
  __shared__ __align__(32) bf16_t bsh[2][1024];   // 2 buffers x (2 fragments)
  const int lane = threadIdx.x;
  const int mt = blockIdx.x, nt = blockIdx.y;
  const int rA = lane & 15, kh2 = lane >> 4;
  const int OHW = OH * OW;
  int p   = mt * 16 + rA;
  int n   = p / OHW; int rem = p - n * OHW;
  int oh  = rem / OW; int ow = rem - oh * OW;
  int ih0 = oh * stride - pad, iw0 = ow * stride - pad;
  const int Kpt = Cin >> 5;            // K-blocks per filter tap (even for all layers)
  const int Kb  = KH * KW * Kpt;       // total K-blocks (even)
  const int NC  = Kb >> 1;             // chunks of 2 K-blocks
  const bf16_t* wg = wpack + ((long)nt * Kb) * 512;   // packed B for this ntile
  const unsigned ldsb = (unsigned)(uintptr_t)&bsh[0][0]; // LDS byte offset (low 32b)

  auto issue_chunk = [&](int c) {
    const char* g = (const char*)(wg + (long)c * 1024);   // 2 KB chunk
    unsigned l = ldsb + (unsigned)((c & 1) * 2048);
#pragma unroll
    for (int i = 0; i < 4; ++i) {                         // 4 x 512B wave-issues
      unsigned long long ga = (unsigned long long)(uintptr_t)(g + i * 512 + lane * 16);
      unsigned la = l + (unsigned)(i * 512 + lane * 16);
      asm volatile("global_load_async_to_lds_b128 %0, %1, off"
                   :: "v"(la), "v"(ga) : "memory");
    }
  };

  v8f acc = {};
  issue_chunk(0);
  for (int c = 0; c < NC; ++c) {
    if (c + 1 < NC) {
      asm volatile("s_wait_dscnt 0x0" ::: "memory");      // WAR: buffer reads done
      issue_chunk(c + 1);
      asm volatile("s_wait_asynccnt 0x4" ::: "memory");   // chunk c resident
    } else {
      asm volatile("s_wait_asynccnt 0x0" ::: "memory");
    }
    const bf16_t* bb = &bsh[c & 1][0];
#pragma unroll
    for (int j = 0; j < 2; ++j) {
      int kb  = c * 2 + j;
      int tap = kb / Kpt, cb = kb - tap * Kpt;
      int ky  = tap / KW, kx = tap - ky * KW;
      int ih  = ih0 + ky, iw = iw0 + kx;
      bool ok = ((unsigned)ih < (unsigned)IH) && ((unsigned)iw < (unsigned)IW);
      v16bf a = {};
      if (ok)
        a = load_afrag(in + (((long)n * IH + ih) * IW + iw) * Cin + cb * 32 + kh2 * 8);
      v16bf b = *(const v16bf*)(bb + j * 512 + lane * 16);  // ds_load from LDS
      acc = wmma_bf16(a, b, acc);                           // EXEC full here
    }
  }
  const int nc = nt * 16 + (lane & 15);
  const float sc_ = scale[nc], bi_ = bias[nc];
#pragma unroll
  for (int v = 0; v < 8; ++v) {
    int m  = v + 8 * kh2;                 // C/D layout: lanes>=16 hold rows 8..15
    int p2 = mt * 16 + m;
    int n2 = p2 / OHW; int r2 = p2 - n2 * OHW;
    int oh2 = r2 / OW; int ow2 = r2 - oh2 * OW;
    long o = (((long)n2 * OH + oh2) * OW + ow2) * Cout + nc;
    float val = acc[v] * sc_ + bi_;
    if (resid) val += bf2f(resid[o]);
    if (relu)  val = fmaxf(val, 0.f);
    out[o] = f2bf(val);
  }
}

// ---------------------------------------------------------------------------
// Global average pool [16,7,7,512] -> feats, initializes both h buffers.
// ---------------------------------------------------------------------------
__global__ void avgpool_init_h(const bf16_t* __restrict__ in,
                               float* __restrict__ hf, bf16_t* __restrict__ hb) {
  int idx = blockIdx.x * blockDim.x + threadIdx.x;
  if (idx >= 16 * 512) return;
  int c = idx & 511, n = idx >> 9;
  float sum = 0.f;
  for (int i = 0; i < 49; ++i) sum += bf2f(in[((long)n * 49 + i) * 512 + c]);
  float v = sum * (1.f / 49.f);
  hf[idx] = v;
  hb[idx] = f2bf(v);
}

__global__ void init_x(const float* __restrict__ E, bf16_t* __restrict__ X) {
  long idx = (long)blockIdx.x * blockDim.x + threadIdx.x; // X [32][16][512]
  if (idx >= 32L * 16 * 512) return;
  int d = (int)(idx & 511);
  long t = idx >> 13;                                     // /(16*512)
  X[idx] = (t == 0) ? f2bf(E[101 * 512 + d]) : f2bf(0.f); // BOS_ID = 101
}

__global__ void set_x(const float* __restrict__ hf, bf16_t* __restrict__ Xrow) {
  int idx = blockIdx.x * blockDim.x + threadIdx.x;
  if (idx < 16 * 512) Xrow[idx] = f2bf(hf[idx]);
}

// ---------------------------------------------------------------------------
// GI[t] = X[t] @ Wih^T + bih : 16x1536, K=512.  One wave per 16x64 tile:
// one A-fragment feeds 4 WMMAs (4x arithmetic intensity on A).
// ---------------------------------------------------------------------------
__global__ __launch_bounds__(32) void gi_wmma(const bf16_t* __restrict__ X,
                                              const bf16_t* __restrict__ wihP,
                                              const float* __restrict__ bih,
                                              float* __restrict__ gi) {
  const int lane = threadIdx.x, nt4 = blockIdx.x;         // 24 blocks
  const int rA = lane & 15, kh2 = lane >> 4;
  const bf16_t* ap = X + rA * 512 + kh2 * 8;
  const bf16_t* wp = wihP + ((long)nt4 * 64) * 512 + (long)lane * 16;
  v8f acc0 = {}, acc1 = {}, acc2 = {}, acc3 = {};
#pragma unroll 2
  for (int kb = 0; kb < 16; ++kb) {
    v16bf a = load_afrag(ap + kb * 32);
    long fo = (long)kb * 512;
    acc0 = wmma_bf16(a, *(const v16bf*)(wp + fo), acc0);
    acc1 = wmma_bf16(a, *(const v16bf*)(wp + 8192 + fo), acc1);
    acc2 = wmma_bf16(a, *(const v16bf*)(wp + 16384 + fo), acc2);
    acc3 = wmma_bf16(a, *(const v16bf*)(wp + 24576 + fo), acc3);
  }
  v8f accs[4] = {acc0, acc1, acc2, acc3};
#pragma unroll
  for (int u = 0; u < 4; ++u) {
    int nc = nt4 * 64 + u * 16 + (lane & 15);
    float bb = bih[nc];
#pragma unroll
    for (int v = 0; v < 8; ++v) gi[(v + 8 * kh2) * 1536 + nc] = accs[u][v] + bb;
  }
}

// ---------------------------------------------------------------------------
// One GRU cell step (latency-critical path): gh = h@Whh^T, then gates.
// Block j owns output columns [16j,16j+16) of h; computes r/z/n tiles together.
// ---------------------------------------------------------------------------
__global__ __launch_bounds__(32) void gru_step(
    const bf16_t* __restrict__ hb_in, const float* __restrict__ hf_in,
    const bf16_t* __restrict__ whhP, const float* __restrict__ bhh,
    const float* __restrict__ gi, float* __restrict__ hf_out,
    bf16_t* __restrict__ hb_out) {
  const int j = blockIdx.x, lane = threadIdx.x;
  const int rA = lane & 15, kh2 = lane >> 4;
  const bf16_t* ap = hb_in + rA * 512 + kh2 * 8;
  const bf16_t* bR = whhP + ((long)(j)      * 16) * 512 + (long)lane * 16;
  const bf16_t* bZ = whhP + ((long)(j + 32) * 16) * 512 + (long)lane * 16;
  const bf16_t* bN = whhP + ((long)(j + 64) * 16) * 512 + (long)lane * 16;
  v8f ar = {}, az = {}, an = {};
#pragma unroll 2
  for (int kb = 0; kb < 16; ++kb) {
    v16bf a = load_afrag(ap + kb * 32);
    long fo = (long)kb * 512;
    ar = wmma_bf16(a, *(const v16bf*)(bR + fo), ar);
    az = wmma_bf16(a, *(const v16bf*)(bZ + fo), az);
    an = wmma_bf16(a, *(const v16bf*)(bN + fo), an);
  }
  int nc = j * 16 + (lane & 15);
  float bhr = bhh[nc], bhz = bhh[512 + nc], bhn = bhh[1024 + nc];
#pragma unroll
  for (int v = 0; v < 8; ++v) {
    int m = v + 8 * kh2;
    float ir  = gi[m * 1536 + nc];
    float iz  = gi[m * 1536 + 512 + nc];
    float in_ = gi[m * 1536 + 1024 + nc];
    float r   = 1.f / (1.f + __expf(-(ir + ar[v] + bhr)));
    float z   = 1.f / (1.f + __expf(-(iz + az[v] + bhz)));
    float ng  = tanhf(in_ + r * (an[v] + bhn));
    float hn2 = (1.f - z) * ng + z * hf_in[m * 512 + nc];
    hf_out[m * 512 + nc] = hn2;
    hb_out[m * 512 + nc] = f2bf(hn2);
  }
}

// ---------------------------------------------------------------------------
// logits[b][t][v] = X[t*16+b] . Wp[v] + bp[v]   (vocab padded 30522 -> 30528)
// 16x64 tile per wave: 1 A-fragment -> 4 WMMAs per K-block.
// ---------------------------------------------------------------------------
__global__ __launch_bounds__(32) void proj_wmma(const bf16_t* __restrict__ X,
                                                const bf16_t* __restrict__ wpP,
                                                const float* __restrict__ bp,
                                                float* __restrict__ logits) {
  const int mt = blockIdx.x, nt4 = blockIdx.y, lane = threadIdx.x; // (32, 477)
  const int rA = lane & 15, kh2 = lane >> 4;
  const bf16_t* ap = X + ((long)mt * 16 + rA) * 512 + kh2 * 8;
  const bf16_t* wp = wpP + ((long)nt4 * 64) * 512 + (long)lane * 16;
  v8f acc0 = {}, acc1 = {}, acc2 = {}, acc3 = {};
#pragma unroll 2
  for (int kb = 0; kb < 16; ++kb) {
    v16bf a = load_afrag(ap + kb * 32);
    long fo = (long)kb * 512;
    acc0 = wmma_bf16(a, *(const v16bf*)(wp + fo), acc0);
    acc1 = wmma_bf16(a, *(const v16bf*)(wp + 8192 + fo), acc1);
    acc2 = wmma_bf16(a, *(const v16bf*)(wp + 16384 + fo), acc2);
    acc3 = wmma_bf16(a, *(const v16bf*)(wp + 24576 + fo), acc3);
  }
  v8f accs[4] = {acc0, acc1, acc2, acc3};
#pragma unroll
  for (int u = 0; u < 4; ++u) {
    int nc = nt4 * 64 + u * 16 + (lane & 15);
    if (nc >= 30522) continue;
    float bb = bp[nc];
#pragma unroll
    for (int v = 0; v < 8; ++v) {
      int p = mt * 16 + v + 8 * kh2;
      int t = p >> 4, b = p & 15;
      logits[((long)b * 32 + t) * 30522 + nc] = accs[u][v] + bb;
    }
  }
}

__global__ __launch_bounds__(256) void softmax_rows(float* __restrict__ logits, int V) {
  __shared__ float red[256];
  float* x = logits + (long)blockIdx.x * V;
  const int tid = threadIdx.x;
  float mx = -INFINITY;
  for (int i = tid; i < V; i += 256) mx = fmaxf(mx, x[i]);
  red[tid] = mx; __syncthreads();
  for (int s = 128; s > 0; s >>= 1) { if (tid < s) red[tid] = fmaxf(red[tid], red[tid + s]); __syncthreads(); }
  mx = red[0]; __syncthreads();
  float sm = 0.f;
  for (int i = tid; i < V; i += 256) { float e = __expf(x[i] - mx); x[i] = e; sm += e; }
  red[tid] = sm; __syncthreads();
  for (int s = 128; s > 0; s >>= 1) { if (tid < s) red[tid] += red[tid + s]; __syncthreads(); }
  float inv = 1.f / red[0];
  for (int i = tid; i < V; i += 256) x[i] *= inv;
}

// ---------------------------------------------------------------------------
// Host orchestration
// ---------------------------------------------------------------------------
static inline unsigned cdiv(long a, long b) { return (unsigned)((a + b - 1) / b); }

extern "C" void kernel_launch(void* const* d_in, const int* in_sizes, int n_in,
                              void* d_out, int out_size, void* d_ws, size_t ws_size,
                              hipStream_t stream) {
  (void)in_sizes; (void)n_in; (void)out_size; (void)ws_size;
  // ---- input map: jax pytree order (dict keys sorted: batch_images, params;
  //      params: blocks, embed, gru, proj, stem; None downsample dropped) ----
  const float* img = (const float*)d_in[0];
  const float *bW1[8], *bs1[8], *bb1[8], *bW2[8], *bs2[8], *bb2[8];
  const float *dW[8], *dsc[8], *dbi[8];
  const int  bc_in[8]  = {64, 64, 64, 128, 128, 256, 256, 512};
  const int  bc_out[8] = {64, 64, 128, 128, 256, 256, 512, 512};
  const bool bdown[8]  = {false, false, true, false, true, false, true, false};
  int ii = 1;
  for (int i = 0; i < 8; ++i) {
    bW1[i] = (const float*)d_in[ii++]; bs1[i] = (const float*)d_in[ii++]; bb1[i] = (const float*)d_in[ii++];
    bW2[i] = (const float*)d_in[ii++]; bs2[i] = (const float*)d_in[ii++]; bb2[i] = (const float*)d_in[ii++];
    if (bdown[i]) { dW[i] = (const float*)d_in[ii++]; dsc[i] = (const float*)d_in[ii++]; dbi[i] = (const float*)d_in[ii++]; }
    else { dW[i] = nullptr; dsc[i] = nullptr; dbi[i] = nullptr; }
  }
  const float* E   = (const float*)d_in[ii++];
  const float* Wih = (const float*)d_in[ii++];
  const float* Whh = (const float*)d_in[ii++];
  const float* bih = (const float*)d_in[ii++];
  const float* bhh = (const float*)d_in[ii++];
  const float* Wp  = (const float*)d_in[ii++];
  const float* bp  = (const float*)d_in[ii++];
  const float* stW = (const float*)d_in[ii++];
  const float* stS = (const float*)d_in[ii++];
  const float* stB = (const float*)d_in[ii++];

  // ---- workspace arena (~120 MB) ----
  size_t off = 0;
  auto arena = [&](size_t bytes) -> void* {
    off = (off + 255) & ~(size_t)255;
    void* p = (char*)d_ws + off;
    off += bytes;
    return p;
  };
  bf16_t* stemOut = (bf16_t*)arena(16L * 112 * 112 * 64 * 2);
  bf16_t* bufA = (bf16_t*)arena(16L * 56 * 56 * 64 * 2);
  bf16_t* bufB = (bf16_t*)arena(16L * 56 * 56 * 64 * 2);
  bf16_t* bufC = (bf16_t*)arena(16L * 56 * 56 * 64 * 2);
  bf16_t* dbuf = (bf16_t*)arena(16L * 28 * 28 * 128 * 2);
  bf16_t* cw1[8], *cw2[8], *cwd[8];
  for (int i = 0; i < 8; ++i) {
    cw1[i] = (bf16_t*)arena((size_t)bc_out[i] * bc_in[i] * 9 * 2);
    cw2[i] = (bf16_t*)arena((size_t)bc_out[i] * bc_out[i] * 9 * 2);
    cwd[i] = bdown[i] ? (bf16_t*)arena((size_t)bc_out[i] * bc_in[i] * 2) : nullptr;
  }
  bf16_t* wihP = (bf16_t*)arena(1536L * 512 * 2);
  bf16_t* whhP = (bf16_t*)arena(1536L * 512 * 2);
  bf16_t* wpP  = (bf16_t*)arena(30528L * 512 * 2);
  float*  hf[2] = {(float*)arena(16L * 512 * 4), (float*)arena(16L * 512 * 4)};
  bf16_t* hb[2] = {(bf16_t*)arena(16L * 512 * 2), (bf16_t*)arena(16L * 512 * 2)};
  bf16_t* X  = (bf16_t*)arena(32L * 16 * 512 * 2);
  float*  GI = (float*)arena(32L * 16 * 1536 * 4);
  float*  outLogits = (float*)d_out;

  // ---- weight packing (re-done each call: deterministic, no cached state) ----
  for (int i = 0; i < 8; ++i) {
    long t1 = (long)bc_out[i] * bc_in[i] * 9;
    pack_conv<<<cdiv(t1, 256), 256, 0, stream>>>(bW1[i], cw1[i], bc_out[i], bc_in[i], 3, 3);
    long t2 = (long)bc_out[i] * bc_out[i] * 9;
    pack_conv<<<cdiv(t2, 256), 256, 0, stream>>>(bW2[i], cw2[i], bc_out[i], bc_out[i], 3, 3);
    if (bdown[i]) {
      long td = (long)bc_out[i] * bc_in[i];
      pack_conv<<<cdiv(td, 256), 256, 0, stream>>>(dW[i], cwd[i], bc_out[i], bc_in[i], 1, 1);
    }
  }
  pack_dense<<<cdiv(1536L * 512, 256), 256, 0, stream>>>(Wih, wihP, 512, 1536, 1536);
  pack_dense<<<cdiv(1536L * 512, 256), 256, 0, stream>>>(Whh, whhP, 512, 1536, 1536);
  pack_dense<<<cdiv(30528L * 512, 256), 256, 0, stream>>>(Wp, wpP, 512, 30528, 30522);

  // ---- ResNet18 ----
  stem_conv<<<cdiv(16L * 112 * 112 * 64, 256), 256, 0, stream>>>(img, stW, stS, stB, stemOut);
  maxpool3x3s2<<<cdiv(16L * 56 * 56 * 64, 256), 256, 0, stream>>>(stemOut, bufA);

  bf16_t *cur = bufA, *y1b = bufB, *outb = bufC;
  int H = 56;
  for (int i = 0; i < 8; ++i) {
    int cin = bc_in[i], cout = bc_out[i];
    int stride = bdown[i] ? 2 : 1;
    int IHW = H, OHd = H / stride;
    long M = 16L * OHd * OHd;
    dim3 g1(cdiv(M, 16), cout / 16);
    conv_bn_wmma<<<g1, 32, 0, stream>>>(cur, cw1[i], bs1[i], bb1[i], nullptr, y1b,
                                        16, IHW, IHW, cin, OHd, OHd, cout, 3, 3,
                                        stride, 1, 1);
    const bf16_t* sc = cur;
    if (bdown[i]) {
      conv_bn_wmma<<<g1, 32, 0, stream>>>(cur, cwd[i], dsc[i], dbi[i], nullptr, dbuf,
                                          16, IHW, IHW, cin, OHd, OHd, cout, 1, 1,
                                          2, 0, 0);
      sc = dbuf;
    }
    conv_bn_wmma<<<g1, 32, 0, stream>>>(y1b, cw2[i], bs2[i], bb2[i], sc, outb,
                                        16, OHd, OHd, cout, OHd, OHd, cout, 3, 3,
                                        1, 1, 1);
    bf16_t* t = cur; cur = outb; outb = y1b; y1b = t;  // rotate buffers
    H = OHd;
  }

  // ---- pooler + GRU init ----
  avgpool_init_h<<<cdiv(16 * 512, 256), 256, 0, stream>>>(cur, hf[0], hb[0]);
  init_x<<<cdiv(32L * 16 * 512, 256), 256, 0, stream>>>(E, X);
  gi_wmma<<<24, 32, 0, stream>>>(X, wihP, bih, GI);  // GI row 0

  // ---- decode: outer t re-runs cell over prefix; GI rows hoisted out ----
  int ch = 0;
  for (int t = 0; t < 31; ++t) {
    for (int s = 0; s <= t; ++s) {
      gru_step<<<32, 32, 0, stream>>>(hb[ch], hf[ch], whhP, bhh,
                                      GI + (long)s * 16 * 1536, hf[1 - ch], hb[1 - ch]);
      ch ^= 1;
    }
    set_x<<<32, 256, 0, stream>>>(hf[ch], X + (long)(t + 1) * 16 * 512);
    if (t + 1 <= 30)
      gi_wmma<<<24, 32, 0, stream>>>(X + (long)(t + 1) * 16 * 512, wihP, bih,
                                     GI + (long)(t + 1) * 16 * 1536);
  }

  // ---- projection + softmax ----
  proj_wmma<<<dim3(32, 30528 / 16 / 4), 32, 0, stream>>>(X, wpP, bp, outLogits);
  softmax_rows<<<16 * 32, 256, 0, stream>>>(outLogits, 30522);
}
